// MambaBlock_72009421684767
// MI455X (gfx1250) — matrix operand order repaired and checked by reading, not compiled, not run on hardware
//
#include <hip/hip_runtime.h>

// Multi-head self-attention block for MI455X (gfx1250, wave32, WMMA).
// Pipeline: fp32->bf16 convert (+weight transpose) -> QKV GEMM (WMMA bf16,
// async-to-LDS double-buffered) -> flash attention (WMMA bf16, fp32 softmax)
// -> out GEMM + bias (fp32 out).

#define D_MODEL 1024
#define N_HEADS 16
#define HD      64
#define SEQ     2048
#define NBATCH  2
#define ROWS    (NBATCH * SEQ)     /* 4096 */
#define ATT_SCALE 0.03125f         /* 1024^-0.5 (reference scales by full dim) */

typedef __attribute__((ext_vector_type(16))) __bf16 v16bf;
typedef __attribute__((ext_vector_type(8)))  float  v8f;

struct __attribute__((aligned(16))) U4 { unsigned int x, y, z, w; };

static __device__ __forceinline__ unsigned short f32_to_bf16(float f) {
  union { float f; unsigned int u; } c; c.f = f;
  unsigned int r = 0x7FFFu + ((c.u >> 16) & 1u);   // round-to-nearest-even
  return (unsigned short)((c.u + r) >> 16);
}

static __device__ __forceinline__ v16bf frag_ld2(const unsigned short* p0,
                                                 const unsigned short* p1) {
  union { U4 u[2]; v16bf v; } f;
  f.u[0] = *(const U4*)p0;
  f.u[1] = *(const U4*)p1;
  return f.v;
}
// A-operand (16xK=32 bf16): lane holds row M=lane%16; halves h -> k = (lane/16)*16 + h
static __device__ __forceinline__ v16bf fragA(const unsigned short* row, int lh8) {
  return frag_ld2(row + 2 * lh8, row + 2 * lh8 + 8);
}
// B-operand (K=32x16 bf16): lane holds col N=lane%16; halves h -> k = (lane/16)*16 + h
// caller pre-offsets p by (lane/16)*16
static __device__ __forceinline__ v16bf fragB(const unsigned short* p) {
  return frag_ld2(p, p + 8);
}

static __device__ __forceinline__ v8f bf16_wmma(v16bf a, v16bf b, v8f c) {
  return __builtin_amdgcn_wmma_f32_16x16x32_bf16(false, a, false, b,
                                                 (short)0, c, false, false);
}

// ---------------------------------------------------------------- converts
__global__ __launch_bounds__(256) void cvt_copy_kernel(
    const float* __restrict__ in, unsigned short* __restrict__ out, int n) {
  int i = blockIdx.x * blockDim.x + threadIdx.x;
  if (i < n) out[i] = f32_to_bf16(in[i]);
}

// in: [K][N] fp32 row-major -> out: [N][K] bf16 (transposed)
__global__ __launch_bounds__(256) void cvt_transpose_kernel(
    const float* __restrict__ in, unsigned short* __restrict__ out, int K, int N) {
  int i = blockIdx.x * blockDim.x + threadIdx.x;
  if (i < K * N) {
    int k = i / N, n = i - k * N;
    out[(size_t)n * K + k] = f32_to_bf16(in[i]);
  }
}

// ------------------------------------------------------------------- GEMM
// C[M][N] = A[M][K] * B[K][N], B supplied transposed as Bt[N][K].
// Block: 128 threads (4 waves) -> 64x128 tile; wave -> 32x64 (2x4 WMMA tiles).
// K staged 64 at a time into double-buffered LDS via async-to-LDS DMA.
__global__ __launch_bounds__(128) void gemm_bf16_wmma(
    const unsigned short* __restrict__ A,
    const unsigned short* __restrict__ Bt,
    unsigned short* __restrict__ Cb,     // bf16 output (or nullptr)
    float* __restrict__ Cf,              // fp32 output (or nullptr)
    const float* __restrict__ bias,      // optional fp32 bias (len N)
    int M, int N, int K) {
  __shared__ __attribute__((aligned(16))) unsigned short sA[2][64 * 64];
  __shared__ __attribute__((aligned(16))) unsigned short sB[2][128 * 64];
  const int tid  = threadIdx.x;
  const int lane = tid & 31;
  const int wave = tid >> 5;
  const int m0 = blockIdx.y * 64;
  const int n0 = blockIdx.x * 128;
  const int wm = (wave >> 1) << 5;     // 0 / 32
  const int wn = (wave & 1) << 6;      // 0 / 64
  const int lm   = lane & 15;
  const int lh8  = (lane >> 4) << 3;
  const int lh16 = (lane >> 4) << 4;

  // staging map: A -> thread copies 32 halves (64B) of the 64x64 tile,
  //              B -> thread copies one full 64-half row (128B) of 128x64 tile
  const int ar = tid >> 1;
  const int ac = (tid & 1) << 5;
  const unsigned short* gA = A  + (size_t)(m0 + ar) * K + ac;
  const unsigned short* gB = Bt + (size_t)(n0 + tid) * K;

  auto stage = [&](int buf, int k0) {
    unsigned la = (unsigned)(uintptr_t)(sA[buf] + ar * 64 + ac);
    unsigned lb = (unsigned)(uintptr_t)(sB[buf] + tid * 64);
    const unsigned short* pa = gA + k0;
    const unsigned short* pb = gB + k0;
    asm volatile(
        "global_load_async_to_lds_b128 %0, %2, off\n\t"
        "global_load_async_to_lds_b128 %0, %2, off offset:16\n\t"
        "global_load_async_to_lds_b128 %0, %2, off offset:32\n\t"
        "global_load_async_to_lds_b128 %0, %2, off offset:48\n\t"
        "global_load_async_to_lds_b128 %1, %3, off\n\t"
        "global_load_async_to_lds_b128 %1, %3, off offset:16\n\t"
        "global_load_async_to_lds_b128 %1, %3, off offset:32\n\t"
        "global_load_async_to_lds_b128 %1, %3, off offset:48\n\t"
        "global_load_async_to_lds_b128 %1, %3, off offset:64\n\t"
        "global_load_async_to_lds_b128 %1, %3, off offset:80\n\t"
        "global_load_async_to_lds_b128 %1, %3, off offset:96\n\t"
        "global_load_async_to_lds_b128 %1, %3, off offset:112"
        :: "v"(la), "v"(lb), "v"(pa), "v"(pb) : "memory");
  };

  v8f acc[2][4];
#pragma unroll
  for (int i = 0; i < 2; ++i)
#pragma unroll
    for (int j = 0; j < 4; ++j) acc[i][j] = (v8f){};

  stage(0, 0);
  int buf = 0;
  for (int k0 = 0; k0 < K; k0 += 64) {
    asm volatile("s_wait_asynccnt 0x0" ::: "memory");  // own tile landed
    __syncthreads();                                   // everyone's tile landed
    if (k0 + 64 < K) stage(buf ^ 1, k0 + 64);          // prefetch next tile
    const unsigned short* cA = sA[buf];
    const unsigned short* cB = sB[buf];
#pragma unroll
    for (int kk = 0; kk < 64; kk += 32) {
      v16bf a0 = fragA(cA + (wm + lm) * 64 + kk, lh8);
      v16bf a1 = fragA(cA + (wm + 16 + lm) * 64 + kk, lh8);
      v16bf b0 = fragB(cB + (wn + lm) * 64 + kk + lh16);
      v16bf b1 = fragB(cB + (wn + 16 + lm) * 64 + kk + lh16);
      v16bf b2 = fragB(cB + (wn + 32 + lm) * 64 + kk + lh16);
      v16bf b3 = fragB(cB + (wn + 48 + lm) * 64 + kk + lh16);
      acc[0][0] = bf16_wmma(a0, b0, acc[0][0]);
      acc[0][1] = bf16_wmma(a0, b1, acc[0][1]);
      acc[0][2] = bf16_wmma(a0, b2, acc[0][2]);
      acc[0][3] = bf16_wmma(a0, b3, acc[0][3]);
      acc[1][0] = bf16_wmma(a1, b0, acc[1][0]);
      acc[1][1] = bf16_wmma(a1, b1, acc[1][1]);
      acc[1][2] = bf16_wmma(a1, b2, acc[1][2]);
      acc[1][3] = bf16_wmma(a1, b3, acc[1][3]);
    }
    buf ^= 1;
  }

  const int hlf = lane >> 4;
#pragma unroll
  for (int i = 0; i < 2; ++i) {
#pragma unroll
    for (int j = 0; j < 4; ++j) {
      const int n = n0 + wn + j * 16 + lm;
      const float bv = (Cf && bias) ? bias[n] : 0.0f;
#pragma unroll
      for (int r = 0; r < 8; ++r) {
        const int m = m0 + wm + i * 16 + r + 8 * hlf;
        const float v = acc[i][j][r];
        if (Cf) Cf[(size_t)m * N + n] = v + bv;
        else    Cb[(size_t)m * N + n] = f32_to_bf16(v);
      }
    }
  }
}

// -------------------------------------------------------- flash attention
// qkv: [4096][3072] bf16, row layout: [Q(1024) | K(1024) | V(1024)],
// each 1024 = head*64 + d.  Output Ab: [4096][1024] bf16 (b, n, h*d).
// Block: 128 threads = 4 waves; block owns (b, h, 64 query rows); each wave
// owns a 16-row query tile and flash-loops over 32-key/value chunks.
__global__ __launch_bounds__(128) void attn_wmma_kernel(
    const unsigned short* __restrict__ qkv,
    unsigned short* __restrict__ Ab) {
  __shared__ __attribute__((aligned(16))) unsigned short sK[32 * 64];   // [j][d]
  __shared__ __attribute__((aligned(16))) unsigned short sVt[64 * 32];  // [d][j]
  __shared__ __attribute__((aligned(16))) unsigned short sP[4][16 * 32];

  const int tid  = threadIdx.x;
  const int lane = tid & 31;
  const int wave = tid >> 5;
  const int blk  = blockIdx.x;
  const int i64  = blk & 31;        // 64-row tile index within sequence
  const int bh   = blk >> 5;
  const int h    = bh & 15;
  const int b    = bh >> 4;
  const int lm   = lane & 15;
  const int hlf  = lane >> 4;
  const int lh8  = hlf << 3;
  const int lh16 = hlf << 4;

  // Q fragments for this wave's 16 rows (A-operand, d in [0,32) and [32,64))
  const size_t rowQ = (size_t)(b * SEQ + i64 * 64 + wave * 16 + lm) * 3072 + h * HD;
  const v16bf qf0 = fragA(qkv + rowQ, lh8);
  const v16bf qf1 = fragA(qkv + rowQ + 32, lh8);

  v8f o0 = {}, o1 = {}, o2 = {}, o3 = {};
  float rmax[8], rsum[8];
#pragma unroll
  for (int r = 0; r < 8; ++r) { rmax[r] = -3.0e38f; rsum[r] = 0.0f; }

  // staging map: thread copies 16 halves of the 32x64 K tile / V tile
  const int jr = (tid * 16) >> 6;   // 0..31
  const int jc = (tid * 16) & 63;   // 0,16,32,48
  const unsigned lk = (unsigned)(uintptr_t)(sK + jr * 64 + jc);

  for (int j0 = 0; j0 < SEQ; j0 += 32) {
    __syncthreads();   // previous tile fully consumed
    const size_t srow = (size_t)(b * SEQ + j0 + jr) * 3072 + h * HD + jc;
    {
      // K tile via async DMA (row-major, 32B contiguous per thread)
      const unsigned short* gk = qkv + srow + 1024;
      asm volatile(
          "global_load_async_to_lds_b128 %0, %1, off\n\t"
          "global_load_async_to_lds_b128 %0, %1, off offset:16"
          :: "v"(lk), "v"(gk) : "memory");
      // V staged transposed: sVt[d][j] (scatter, must go through VGPRs)
      union { U4 u[2]; unsigned short s[16]; } vv;
      const U4* gv = (const U4*)(qkv + srow + 2048);
      vv.u[0] = gv[0]; vv.u[1] = gv[1];
#pragma unroll
      for (int q = 0; q < 16; ++q) sVt[(jc + q) * 32 + jr] = vv.s[q];
    }
    asm volatile("s_wait_asynccnt 0x0" ::: "memory");
    __syncthreads();

    // S = Q * K^T  (two 16x16 column tiles, K-dim = 64 = 2 WMMA steps)
    v8f s0 = {}, s1 = {};
    s0 = bf16_wmma(qf0, fragB(sK + lm * 64 + 0  + lh16), s0);
    s0 = bf16_wmma(qf1, fragB(sK + lm * 64 + 32 + lh16), s0);
    s1 = bf16_wmma(qf0, fragB(sK + (16 + lm) * 64 + 0  + lh16), s1);
    s1 = bf16_wmma(qf1, fragB(sK + (16 + lm) * 64 + 32 + lh16), s1);

    // online softmax; stage P (bf16, A-operand layout) via per-wave LDS
    unsigned short* Pw = sP[wave];
#pragma unroll
    for (int r = 0; r < 8; ++r) {
      float x0 = s0[r] * ATT_SCALE;
      float x1 = s1[r] * ATT_SCALE;
      float mx = fmaxf(x0, x1);
      mx = fmaxf(mx, __shfl_xor(mx, 1));
      mx = fmaxf(mx, __shfl_xor(mx, 2));
      mx = fmaxf(mx, __shfl_xor(mx, 4));
      mx = fmaxf(mx, __shfl_xor(mx, 8));
      const float mnew  = fmaxf(rmax[r], mx);
      const float alpha = __expf(rmax[r] - mnew);
      rmax[r] = mnew;
      const float p0 = __expf(x0 - mnew);
      const float p1 = __expf(x1 - mnew);
      float rs = p0 + p1;
      rs += __shfl_xor(rs, 1);
      rs += __shfl_xor(rs, 2);
      rs += __shfl_xor(rs, 4);
      rs += __shfl_xor(rs, 8);
      rsum[r] = rsum[r] * alpha + rs;
      o0[r] *= alpha; o1[r] *= alpha; o2[r] *= alpha; o3[r] *= alpha;
      const int prow = r + 8 * hlf;                 // C-layout row
      Pw[prow * 32 + lm]      = f32_to_bf16(p0);
      Pw[prow * 32 + 16 + lm] = f32_to_bf16(p1);
    }
    asm volatile("s_wait_dscnt 0x0" ::: "memory");  // intra-wave LDS RAW

    // O += P * V   (A = P 16x32, B = Vt column tiles of 16 d's)
    const v16bf pf = fragA(Pw + lm * 32, lh8);
    o0 = bf16_wmma(pf, fragB(sVt + (0  + lm) * 32 + lh16), o0);
    o1 = bf16_wmma(pf, fragB(sVt + (16 + lm) * 32 + lh16), o1);
    o2 = bf16_wmma(pf, fragB(sVt + (32 + lm) * 32 + lh16), o2);
    o3 = bf16_wmma(pf, fragB(sVt + (48 + lm) * 32 + lh16), o3);
  }

  // normalize and store bf16 [4096][1024]
#pragma unroll
  for (int r = 0; r < 8; ++r) {
    const float inv = 1.0f / rsum[r];
    const int m = i64 * 64 + wave * 16 + r + 8 * hlf;
    const size_t orow = (size_t)(b * SEQ + m) * D_MODEL + h * HD;
    Ab[orow + 0  + lm] = f32_to_bf16(o0[r] * inv);
    Ab[orow + 16 + lm] = f32_to_bf16(o1[r] * inv);
    Ab[orow + 32 + lm] = f32_to_bf16(o2[r] * inv);
    Ab[orow + 48 + lm] = f32_to_bf16(o3[r] * inv);
  }
}

// ------------------------------------------------------------------ launch
extern "C" void kernel_launch(void* const* d_in, const int* in_sizes, int n_in,
                              void* d_out, int out_size, void* d_ws, size_t ws_size,
                              hipStream_t stream) {
  const float* x     = (const float*)d_in[0];   // [2,2048,1024]
  const float* w_qkv = (const float*)d_in[1];   // [1024,3072]
  const float* w_out = (const float*)d_in[2];   // [1024,1024]
  const float* b_out = (const float*)d_in[3];   // [1024]
  float* out = (float*)d_out;                   // [2,2048,1024] fp32

  char* ws = (char*)d_ws;
  unsigned short* Xb    = (unsigned short*)(ws + 0);          // 4096x1024 bf16  (8 MiB)
  unsigned short* WqkvT = (unsigned short*)(ws + 8388608);    // 3072x1024 bf16  (6 MiB)
  unsigned short* WoutT = (unsigned short*)(ws + 14680064);   // 1024x1024 bf16  (2 MiB)
  unsigned short* QKVb  = (unsigned short*)(ws + 16777216);   // 4096x3072 bf16 (24 MiB)
  unsigned short* Ab    = (unsigned short*)(ws + 41943040);   // 4096x1024 bf16  (8 MiB)

  // fp32 -> bf16 conversions (weights transposed for B-operand staging)
  cvt_copy_kernel<<<(ROWS * D_MODEL + 255) / 256, 256, 0, stream>>>(x, Xb, ROWS * D_MODEL);
  cvt_transpose_kernel<<<(D_MODEL * 3 * D_MODEL + 255) / 256, 256, 0, stream>>>(
      w_qkv, WqkvT, D_MODEL, 3 * D_MODEL);
  cvt_transpose_kernel<<<(D_MODEL * D_MODEL + 255) / 256, 256, 0, stream>>>(
      w_out, WoutT, D_MODEL, D_MODEL);

  // QKV projection: [4096,1024] x [1024,3072] -> bf16 [4096,3072]
  gemm_bf16_wmma<<<dim3(3 * D_MODEL / 128, ROWS / 64), 128, 0, stream>>>(
      Xb, WqkvT, QKVb, nullptr, nullptr, ROWS, 3 * D_MODEL, D_MODEL);

  // flash attention: 2 batches * 16 heads * 32 row-tiles = 1024 blocks
  attn_wmma_kernel<<<NBATCH * N_HEADS * (SEQ / 64), 128, 0, stream>>>(QKVb, Ab);

  // output projection + bias: [4096,1024] x [1024,1024] -> fp32 d_out
  gemm_bf16_wmma<<<dim3(D_MODEL / 128, ROWS / 64), 128, 0, stream>>>(
      Ab, WoutT, nullptr, out, b_out, ROWS, D_MODEL, D_MODEL);
}
